// HVQ_TR_switch_27797028340227
// MI455X (gfx1250) — matrix operand order repaired and checked by reading, not compiled, not run on hardware
//
#include <hip/hip_runtime.h>

typedef __attribute__((ext_vector_type(16))) _Float16 v16h;
typedef __attribute__((ext_vector_type(8)))  float    v8f;
typedef __attribute__((ext_vector_type(4)))  float    v4f;

#define DEVINL __device__ __forceinline__

// ---------------------------------------------------------------------------
// constants
// ---------------------------------------------------------------------------
#define BB 16
#define HH 14
#define LL 196          // 14*14 valid rows
#define LP 208          // rows padded to 13 full 16-row tiles (stores unguarded)
#define KK 512
#define CC 272
#define OUT_R 64

__device__ __constant__ int d_c0s[4]  = {0, 24, 56, 112};   // split channel starts
__device__ __constant__ int d_cs[4]   = {24, 32, 56, 160};  // split widths
__device__ __constant__ int d_cpad[4] = {32, 32, 64, 160};  // K padded to mult of 32
__device__ __constant__ int d_cum[4]  = {0, 32, 64, 128};   // cumulative cpad (total 288)

// ---------------------------------------------------------------------------
// CDNA5 WMMA tile helpers (wave32, v_wmma_f32_16x16x32_f16) — branchless loads.
// A 16x32 (MxK) f16, row-major K-contiguous (stride rs, 16-float aligned):
//   lane 0-15: M=lane, K {kb..kb+7} in VGPR0..3, {kb+16..kb+23} in VGPR4..7, kb=k0
//   lane 16-31: same rows, kb=k0+8
// B 32x16 (KxN) f16, stored N-major K-contiguous (stride ns):
//   lane 0-15: N=lane, K {kb..kb+7} then {kb+16..kb+23}, kb=k0; lanes 16-31: kb=k0+8
// C/D 16x16 f32: row = v + 8*(lane>=16), col = lane&15.
// Out-of-range rows/cols are CLAMPED on load (garbage stays in padded D rows).
// ---------------------------------------------------------------------------
DEVINL v16h load_a16(const float* __restrict__ A, int row0, int k0, long rs, int mMax) {
  int lane = threadIdx.x & 31;
  int m = row0 + (lane & 15);
  if (m >= mMax) m = mMax - 1;
  int kb = k0 + ((lane >> 4) << 3);
  const float* base = A + (long)m * rs + kb;
  v4f p0 = *(const v4f*)(base);
  v4f p1 = *(const v4f*)(base + 4);
  v4f p2 = *(const v4f*)(base + 16);
  v4f p3 = *(const v4f*)(base + 20);
  v16h o;
  o[0] = (_Float16)p0[0];  o[1] = (_Float16)p0[1];  o[2] = (_Float16)p0[2];  o[3] = (_Float16)p0[3];
  o[4] = (_Float16)p1[0];  o[5] = (_Float16)p1[1];  o[6] = (_Float16)p1[2];  o[7] = (_Float16)p1[3];
  o[8] = (_Float16)p2[0];  o[9] = (_Float16)p2[1];  o[10] = (_Float16)p2[2]; o[11] = (_Float16)p2[3];
  o[12] = (_Float16)p3[0]; o[13] = (_Float16)p3[1]; o[14] = (_Float16)p3[2]; o[15] = (_Float16)p3[3];
  return o;
}

DEVINL v16h load_b16(const float* __restrict__ Bm, int n0, int k0, long ns, int nMax) {
  int lane = threadIdx.x & 31;
  int n = n0 + (lane & 15);
  if (n >= nMax) n = nMax - 1;
  int kb = k0 + ((lane >> 4) << 3);
  const float* base = Bm + (long)n * ns + kb;
  v4f p0 = *(const v4f*)(base);
  v4f p1 = *(const v4f*)(base + 4);
  v4f p2 = *(const v4f*)(base + 16);
  v4f p3 = *(const v4f*)(base + 20);
  v16h o;
  o[0] = (_Float16)p0[0];  o[1] = (_Float16)p0[1];  o[2] = (_Float16)p0[2];  o[3] = (_Float16)p0[3];
  o[4] = (_Float16)p1[0];  o[5] = (_Float16)p1[1];  o[6] = (_Float16)p1[2];  o[7] = (_Float16)p1[3];
  o[8] = (_Float16)p2[0];  o[9] = (_Float16)p2[1];  o[10] = (_Float16)p2[2]; o[11] = (_Float16)p2[3];
  o[12] = (_Float16)p3[0]; o[13] = (_Float16)p3[1]; o[14] = (_Float16)p3[2]; o[15] = (_Float16)p3[3];
  return o;
}

// ---------------------------------------------------------------------------
// kernels
// ---------------------------------------------------------------------------

__global__ void zero_accum_kernel(float* __restrict__ accum) {
  if (threadIdx.x < 2) accum[threadIdx.x] = 0.f;
}

// cbT[b][n][t] = CB[labels[b]][t][n]   (K-contiguous codebook transpose)
__global__ void pack_cbT_kernel(const float* __restrict__ CB, const int* __restrict__ labels,
                                float* __restrict__ cbT, int d) {
  int idx = blockIdx.x * blockDim.x + threadIdx.x;
  if (idx >= BB * KK * d) return;
  int t = idx % d;
  int r = idx / d;
  int n = r % KK, b = r / KK;
  cbT[idx] = CB[(long)labels[b] * d * KK + (long)t * KK + n];
}

// cbnorm[b][k] = ||cbT[b][k][:]||^2
__global__ void cbnorm_kernel(const float* __restrict__ cbT, float* __restrict__ out, int d) {
  int idx = blockIdx.x * blockDim.x + threadIdx.x;
  if (idx >= BB * KK) return;
  const float* row = cbT + (long)idx * d;
  float s = 0.f;
  for (int t = 0; t < d; ++t) { float v = row[t]; s += v * v; }
  out[idx] = s;
}

// scores[b][m][k] = X[b][m][:] . cbT[b][k][:]   (WMMA; scores padded to LP rows;
// N=512 tile-exact and rows padded -> fully unguarded stores)
template <int D>
__global__ void vq_scores_kernel(const float* __restrict__ X, const float* __restrict__ cbT,
                                 float* __restrict__ scores) {
  int b  = blockIdx.z;
  int m0 = blockIdx.y * 16;
  int nt = blockIdx.x * (blockDim.x >> 5) + (threadIdx.x >> 5);
  if (nt >= KK / 16) return;
  int n0 = nt * 16;
  const float* A  = X   + (long)b * LL * D;
  const float* Bm = cbT + (long)b * KK * D;
  v8f acc = {};
#pragma unroll
  for (int k0 = 0; k0 < D; k0 += 32) {
    v16h a  = load_a16(A,  m0, k0, D, LL);
    v16h bt = load_b16(Bm, n0, k0, D, KK);
    acc = __builtin_amdgcn_wmma_f32_16x16x32_f16(false, a, false, bt, (short)0, acc, false, false);
  }
  int lane = threadIdx.x & 31;
  int n  = n0 + (lane & 15);
  int mb = m0 + ((lane >> 4) << 3);
  float* Dst = scores + (long)b * LP * KK;
#pragma unroll
  for (int v = 0; v < 8; ++v) Dst[(long)(mb + v) * KK + n] = acc[v];
}

// one warp per (b,l): argmin_k (cbnorm[k] - 2*score); gather q from cbT (contiguous);
// write x43 = [feat_l3 | q]; accumulate sum (q - feat_l4)^2
__global__ void argmin1_kernel(const float* __restrict__ scores, const float* __restrict__ cbn,
                               const float* __restrict__ cbT,
                               const float* __restrict__ feat_l4, const float* __restrict__ feat_l3,
                               float* __restrict__ x43, float* __restrict__ accum) {
  int warp = (blockIdx.x * blockDim.x + threadIdx.x) >> 5;
  int lane = threadIdx.x & 31;
  if (warp >= BB * LL) return;
  int b = warp / LL, l = warp % LL;
  const float* sc = scores + ((long)b * LP + l) * KK;
  const float* cn = cbn + (long)b * KK;
  float best = 3.4e38f; int bidx = 0x7fffffff;
  for (int k = lane; k < KK; k += 32) {
    float v = cn[k] - 2.f * sc[k];
    if (v < best) { best = v; bidx = k; }
  }
  for (int off = 16; off; off >>= 1) {
    float ov = __shfl_xor(best, off, 32);
    int   oi = __shfl_xor(bidx, off, 32);
    if (ov < best || (ov == best && oi < bidx)) { best = ov; bidx = oi; }
  }
  const float* q  = cbT + ((long)b * KK + bidx) * 64;
  const float* x  = feat_l4 + ((long)b * LL + l) * 64;
  const float* f3 = feat_l3 + ((long)b * LL + l) * 64;
  float* xo = x43 + ((long)b * LL + l) * 128;
  float locsum = 0.f;
  for (int dd = lane; dd < 64; dd += 32) {
    float qv = q[dd];
    float dv = qv - x[dd];
    locsum += dv * dv;
    xo[64 + dd] = qv;
    xo[dd]      = f3[dd];
  }
  for (int off = 16; off; off >>= 1) locsum += __shfl_xor(locsum, off, 32);
  if (lane == 0) atomicAdd(&accum[0], locsum);
}

// one warp per (b,l): argmin over codebook4; gather q -> dec; accumulate (q - x43)^2
__global__ void argmin2_kernel(const float* __restrict__ scores, const float* __restrict__ cbn,
                               const float* __restrict__ cbT,
                               const float* __restrict__ x43, float* __restrict__ dec,
                               float* __restrict__ accum) {
  int warp = (blockIdx.x * blockDim.x + threadIdx.x) >> 5;
  int lane = threadIdx.x & 31;
  if (warp >= BB * LL) return;
  int b = warp / LL, l = warp % LL;
  const float* sc = scores + ((long)b * LP + l) * KK;
  const float* cn = cbn + (long)b * KK;
  float best = 3.4e38f; int bidx = 0x7fffffff;
  for (int k = lane; k < KK; k += 32) {
    float v = cn[k] - 2.f * sc[k];
    if (v < best) { best = v; bidx = k; }
  }
  for (int off = 16; off; off >>= 1) {
    float ov = __shfl_xor(best, off, 32);
    int   oi = __shfl_xor(bidx, off, 32);
    if (ov < best || (ov == best && oi < bidx)) { best = ov; bidx = oi; }
  }
  const float* q = cbT + ((long)b * KK + bidx) * 128;
  const float* x = x43 + ((long)b * LL + l) * 128;
  float* qo = dec + ((long)b * LL + l) * 128;
  float locsum = 0.f;
  for (int dd = lane; dd < 128; dd += 32) {
    float qv = q[dd];
    float dv = qv - x[dd];
    locsum += dv * dv;
    qo[dd] = qv;
  }
  for (int off = 16; off; off >>= 1) locsum += __shfl_xor(locsum, off, 32);
  if (lane == 0) atomicAdd(&accum[1], locsum);
}

// rec[b][m][c] = dec[b][m][:] . W[labels[b]][c][:] + bias   (WMMA; rec padded to LP
// rows; N=272=17*16 tile-exact -> fully unguarded stores)
__global__ void rec_kernel(const float* __restrict__ dec, const float* __restrict__ W,
                           const float* __restrict__ bias, const int* __restrict__ labels,
                           float* __restrict__ rec) {
  int b  = blockIdx.z;
  int m0 = blockIdx.y * 16;
  int nt = blockIdx.x * (blockDim.x >> 5) + (threadIdx.x >> 5);
  if (nt >= 17) return;
  int n0 = nt * 16;
  int lab = labels[b];
  const float* A  = dec + (long)b * LL * 128;
  const float* Bw = W + (long)lab * CC * 128;
  v8f acc = {};
#pragma unroll
  for (int k0 = 0; k0 < 128; k0 += 32) {
    v16h a  = load_a16(A,  m0, k0, 128, LL);
    v16h bt = load_b16(Bw, n0, k0, 128, CC);
    acc = __builtin_amdgcn_wmma_f32_16x16x32_f16(false, a, false, bt, (short)0, acc, false, false);
  }
  int lane = threadIdx.x & 31;
  int n  = n0 + (lane & 15);
  int mb = m0 + ((lane >> 4) << 3);
  float bv = bias[(long)lab * CC + n];
  float* R = rec + (long)b * LP * CC;
#pragma unroll
  for (int v = 0; v < 8; ++v) R[(long)(mb + v) * CC + n] = acc[v] + bv;
}

// scatter rec[b][l][c] into dino0..3 (concatenated region, [B,196,c_s] each)
__global__ void dino_kernel(const float* __restrict__ rec, float* __restrict__ dino) {
  int idx = blockIdx.x * blockDim.x + threadIdx.x;
  if (idx >= BB * LL * CC) return;
  int c  = idx % CC;
  int bl = idx / CC;
  int b = bl / LL, l = bl % LL;
  int s = (c < 24) ? 0 : (c < 56) ? 1 : (c < 112) ? 2 : 3;
  const long doff[4] = {0L, (long)BB * LL * 24, (long)BB * LL * 56, (long)BB * LL * 112};
  dino[doff[s] + (long)bl * d_cs[s] + (c - d_c0s[s])] = rec[((long)b * LP + l) * CC + c];
}

// inv_org/inv_rec[s][b][l] = 1/(norm + 1e-8) over channels of each split
__global__ void invnorm_kernel(const float* __restrict__ org, const float* __restrict__ rec,
                               float* __restrict__ inv_org, float* __restrict__ inv_rec) {
  int idx = blockIdx.x * blockDim.x + threadIdx.x;
  if (idx >= 4 * BB * LL) return;
  int s = idx / (BB * LL);
  int r = idx % (BB * LL);
  int b = r / LL, l = r % LL;
  int c0 = d_c0s[s], c = d_cs[s];
  const float* o  = org + ((long)b * CC + c0) * LL + l;
  const float* rr = rec + ((long)b * LP + l) * CC + c0;
  float so = 0.f, sr = 0.f;
  for (int t = 0; t < c; ++t) {
    float a = o[(long)t * LL]; so += a * a;
    float bv = rr[t];          sr += bv * bv;
  }
  inv_org[idx] = 1.f / (sqrtf(so) + 1e-8f);
  inv_rec[idx] = 1.f / (sqrtf(sr) + 1e-8f);
}

// pack zero-padded K-contiguous tiles for the similarity GEMMs:
//   packA[s][b][i][t] = org[b][c0s+t][i]   (t < c_s, else 0)
//   packB[s][b][j][t] = rec[b][j][c0s+t]   (t < c_s, else 0)
__global__ void pack_omc_kernel(const float* __restrict__ org, const float* __restrict__ rec,
                                float* __restrict__ packA, float* __restrict__ packB) {
  int idx = blockIdx.x * blockDim.x + threadIdx.x;
  if (idx >= BB * LL * 288) return;
  int t  = idx % 288;
  int bl = idx / 288;
  int s = (t < 32) ? 0 : (t < 64) ? 1 : (t < 128) ? 2 : 3;
  int tt = t - d_cum[s];
  int b = bl / LL, l = bl % LL;
  float va = 0.f, vb = 0.f;
  if (tt < d_cs[s]) {
    int ch = d_c0s[s] + tt;
    va = org[((long)b * CC + ch) * LL + l];
    vb = rec[((long)b * LP + l) * CC + ch];
  }
  long o = (long)BB * LL * d_cum[s] + ((long)b * LL + l) * d_cpad[s] + tt;
  packA[o] = va;
  packB[o] = vb;
}

// omc[s][b][i][j] = 1 - (packA_i . packB_j) * inv_org[i] * inv_rec[j]
// (WMMA, K = CPAD; omc padded to LP rows -> only the column guard remains)
template <int CPAD>
__global__ void omc_kernel(const float* __restrict__ packA, const float* __restrict__ packB,
                           const float* __restrict__ inv_org, const float* __restrict__ inv_rec,
                           float* __restrict__ omc, int s) {
  int b  = blockIdx.z;
  int m0 = blockIdx.y * 16;
  int nt = blockIdx.x * (blockDim.x >> 5) + (threadIdx.x >> 5);
  if (nt >= 13) return;
  int n0 = nt * 16;
  long base = (long)BB * LL * d_cum[s] + (long)b * LL * CPAD;
  const float* A  = packA + base;
  const float* Bm = packB + base;
  v8f acc = {};
#pragma unroll
  for (int k0 = 0; k0 < CPAD; k0 += 32) {
    v16h a  = load_a16(A,  m0, k0, CPAD, LL);
    v16h bt = load_b16(Bm, n0, k0, CPAD, LL);
    acc = __builtin_amdgcn_wmma_f32_16x16x32_f16(false, a, false, bt, (short)0, acc, false, false);
  }
  int lane = threadIdx.x & 31;
  int n  = n0 + (lane & 15);
  int mb = m0 + ((lane >> 4) << 3);
  const float* io = inv_org + ((long)s * BB + b) * LL;
  const float* ir = inv_rec + ((long)s * BB + b) * LL;
  float* O = omc + ((long)(s * BB + b) * LP) * LL;
  if (n < LL) {                      // single uniform-per-lane guard
    float irn = ir[n];
#pragma unroll
    for (int v = 0; v < 8; ++v) {
      int m = mb + v;
      float iom = io[(m < LL) ? m : (LL - 1)];
      O[(long)m * LL + n] = 1.f - acc[v] * iom * irn;
    }
  }
}

// mins[s][b][i] = min_j omc[s][b][i][j]   (one warp per row)
__global__ void minrow_kernel(const float* __restrict__ omc, float* __restrict__ mins) {
  int warp = (blockIdx.x * blockDim.x + threadIdx.x) >> 5;
  int lane = threadIdx.x & 31;
  if (warp >= 4 * BB * LL) return;
  int sb = warp / LL, i = warp % LL;
  const float* row = omc + ((long)sb * LP + i) * LL;
  float best = 3.4e38f;
  for (int j = lane; j < LL; j += 32) best = fminf(best, row[j]);
  for (int off = 16; off; off >>= 1) best = fminf(best, __shfl_xor(best, off, 32));
  if (lane == 0) mins[warp] = best;
}

// pred1[b][l] = sqrt(sum_c (rec[b][l][c] - org[b][c][l])^2)
__global__ void pred1_kernel(const float* __restrict__ org, const float* __restrict__ rec,
                             float* __restrict__ pred1) {
  int idx = blockIdx.x * blockDim.x + threadIdx.x;
  if (idx >= BB * LL) return;
  int b = idx / LL, l = idx % LL;
  const float* o = org + (long)b * CC * LL + l;
  const float* r = rec + ((long)b * LP + l) * CC;
  float s = 0.f;
  for (int c = 0; c < CC; ++c) { float d = r[c] - o[(long)c * LL]; s += d * d; }
  pred1[idx] = sqrtf(s);
}

DEVINL void bilin_setup(int o, int& i0, int& i1, float& w) {
  float f = o * (13.f / 63.f);          // align_corners: (H-1)/(OUT-1)
  i0 = (int)f;
  i1 = i0 + 1; if (i1 > 13) i1 = 13;
  w = f - (float)i0;
}

// a[b][j][s][oy][ox] = bilinear over (iy,ix) of omc[s][b][iy*14+ix][j]
__global__ void upsample_a_kernel(const float* __restrict__ omc, float* __restrict__ a) {
  long idx = (long)blockIdx.x * blockDim.x + threadIdx.x;
  const long total = (long)BB * LL * 4 * OUT_R * OUT_R;
  if (idx >= total) return;
  int ox = (int)(idx & 63);
  int oy = (int)((idx >> 6) & 63);
  long t = idx >> 12;
  int s = (int)(t & 3);
  long bj = t >> 2;
  int j = (int)(bj % LL);
  int b = (int)(bj / LL);
  int y0, y1, x0, x1; float wy, wx;
  bilin_setup(oy, y0, y1, wy);
  bilin_setup(ox, x0, x1, wx);
  const float* O = omc + ((long)(s * BB + b) * LP) * LL + j;
  float v00 = O[(long)(y0 * HH + x0) * LL];
  float v01 = O[(long)(y0 * HH + x1) * LL];
  float v10 = O[(long)(y1 * HH + x0) * LL];
  float v11 = O[(long)(y1 * HH + x1) * LL];
  float v0 = v00 * (1.f - wy) + v10 * wy;
  float v1 = v01 * (1.f - wy) + v11 * wy;
  a[idx] = v0 * (1.f - wx) + v1 * wx;
}

// minsim[b][s][oy][ox]: s>0 -> upsampled mins; s=0 -> mean_s + 0.1*up(pred1)/170
__global__ void minsim_kernel(const float* __restrict__ mins, const float* __restrict__ pred1,
                              float* __restrict__ ms) {
  int idx = blockIdx.x * blockDim.x + threadIdx.x;
  if (idx >= BB * OUT_R * OUT_R) return;
  int ox = idx & 63, oy = (idx >> 6) & 63, b = idx >> 12;
  int y0, y1, x0, x1; float wy, wx;
  bilin_setup(oy, y0, y1, wy);
  bilin_setup(ox, x0, x1, wx);
  float w00 = (1.f - wy) * (1.f - wx), w01 = (1.f - wy) * wx;
  float w10 = wy * (1.f - wx),         w11 = wy * wx;
  int i00 = y0 * HH + x0, i01 = y0 * HH + x1, i10 = y1 * HH + x0, i11 = y1 * HH + x1;
  float msv[4]; float meanv = 0.f;
#pragma unroll
  for (int s = 0; s < 4; ++s) {
    const float* g = mins + ((long)s * BB + b) * LL;
    float v = g[i00] * w00 + g[i01] * w01 + g[i10] * w10 + g[i11] * w11;
    msv[s] = v; meanv += v;
  }
  meanv *= 0.25f;
  const float* p = pred1 + (long)b * LL;
  float pv = p[i00] * w00 + p[i01] * w01 + p[i10] * w10 + p[i11] * w11;
  float* o = ms + (long)b * 4 * (OUT_R * OUT_R) + (long)oy * OUT_R + ox;
  o[0]                 = meanv + 0.1f * pv / 170.f;
  o[1 * OUT_R * OUT_R] = msv[1];
  o[2 * OUT_R * OUT_R] = msv[2];
  o[3 * OUT_R * OUT_R] = msv[3];
}

__global__ void diff_kernel(const float* __restrict__ accum, float* __restrict__ out) {
  if (blockIdx.x == 0 && threadIdx.x == 0)
    out[0] = accum[0] / (float)(LL * 64) + accum[1] / (float)(LL * 128);
}

// ---------------------------------------------------------------------------
// launch
// ---------------------------------------------------------------------------
extern "C" void kernel_launch(void* const* d_in, const int* in_sizes, int n_in,
                              void* d_out, int out_size, void* d_ws, size_t ws_size,
                              hipStream_t stream) {
  (void)in_sizes; (void)n_in; (void)out_size; (void)ws_size;
  const float* org     = (const float*)d_in[0];   // [16,272,14,14]
  const float* feat_l4 = (const float*)d_in[1];   // [16,196,64]
  const float* feat_l3 = (const float*)d_in[2];   // [16,196,64]
  const float* cb5     = (const float*)d_in[3];   // [15,64,512]
  const float* cb4     = (const float*)d_in[4];   // [15,128,512]
  const float* proj_w  = (const float*)d_in[5];   // [15,272,128]
  const float* proj_b  = (const float*)d_in[6];   // [15,272]
  const int*   labels  = (const int*)d_in[7];     // [16]

  float* out = (float*)d_out;
  const long A_ELEMS    = (long)BB * LL * 4 * OUT_R * OUT_R;   // 51,380,224
  const long MS_ELEMS   = (long)BB * 4 * OUT_R * OUT_R;        // 262,144
  const long DINO_ELEMS = (long)BB * LL * CC;                  // 852,992
  float* out_a    = out;
  float* out_ms   = out + A_ELEMS;
  float* out_dino = out + A_ELEMS + MS_ELEMS;
  float* out_diff = out + A_ELEMS + MS_ELEMS + DINO_ELEMS;

  // workspace (floats); two overlaid regions with disjoint lifetimes.
  float* ws = (float*)d_ws;
  float* dec     = ws;                           // 401,408
  float* rec     = dec + 401408;                 // 905,216  = 16*208*272 (padded rows)
  float* inv_org = rec + 905216;                 // 12,544
  float* inv_rec = inv_org + 12544;              // 12,544
  float* mins    = inv_rec + 12544;              // 12,544
  float* pred1   = mins + 12544;                 // 3,136
  float* accum   = pred1 + 3136;                 // 8 (padded, keeps 32B alignment)
  float* uni     = accum + 8;
  // region Y (VQ): cbT 1,048,576 | cbn 8,192 | scores 1,703,936 (16*208*512) | x43 401,408
  float* cbT    = uni;
  float* cbn    = cbT + 1048576;
  float* scores = cbn + 8192;
  float* x43    = scores + 1703936;
  // region X (similarity): packA 903,168 | packB 903,168 | omc 2,609,152 (4*16*208*196)
  float* packA = uni;
  float* packB = packA + 903168;
  float* omc   = packB + 903168;

  zero_accum_kernel<<<1, 32, 0, stream>>>(accum);

  // ---- VQ stage 1 (d=64) ----
  pack_cbT_kernel<<<(BB * KK * 64 + 255) / 256, 256, 0, stream>>>(cb5, labels, cbT, 64);
  cbnorm_kernel<<<(BB * KK + 255) / 256, 256, 0, stream>>>(cbT, cbn, 64);
  vq_scores_kernel<64><<<dim3(8, 13, BB), 128, 0, stream>>>(feat_l4, cbT, scores);
  argmin1_kernel<<<(BB * LL * 32 + 255) / 256, 256, 0, stream>>>(
      scores, cbn, cbT, feat_l4, feat_l3, x43, accum);

  // ---- VQ stage 2 (d=128) ----
  pack_cbT_kernel<<<(BB * KK * 128 + 255) / 256, 256, 0, stream>>>(cb4, labels, cbT, 128);
  cbnorm_kernel<<<(BB * KK + 255) / 256, 256, 0, stream>>>(cbT, cbn, 128);
  vq_scores_kernel<128><<<dim3(8, 13, BB), 128, 0, stream>>>(x43, cbT, scores);
  argmin2_kernel<<<(BB * LL * 32 + 255) / 256, 256, 0, stream>>>(
      scores, cbn, cbT, x43, dec, accum);

  // ---- projection + dino outputs ----
  rec_kernel<<<dim3(5, 13, BB), 128, 0, stream>>>(dec, proj_w, proj_b, labels, rec);
  dino_kernel<<<(int)((BB * (long)LL * CC + 255) / 256), 256, 0, stream>>>(rec, out_dino);

  // ---- cosine-similarity maps ----
  invnorm_kernel<<<(4 * BB * LL + 255) / 256, 256, 0, stream>>>(org, rec, inv_org, inv_rec);
  pack_omc_kernel<<<(BB * LL * 288 + 255) / 256, 256, 0, stream>>>(org, rec, packA, packB);
  omc_kernel<32><<<dim3(4, 13, BB), 128, 0, stream>>>(packA, packB, inv_org, inv_rec, omc, 0);
  omc_kernel<32><<<dim3(4, 13, BB), 128, 0, stream>>>(packA, packB, inv_org, inv_rec, omc, 1);
  omc_kernel<64><<<dim3(4, 13, BB), 128, 0, stream>>>(packA, packB, inv_org, inv_rec, omc, 2);
  omc_kernel<160><<<dim3(4, 13, BB), 128, 0, stream>>>(packA, packB, inv_org, inv_rec, omc, 3);

  // ---- reductions ----
  minrow_kernel<<<(4 * BB * LL * 32 + 255) / 256, 256, 0, stream>>>(omc, mins);
  pred1_kernel<<<(BB * LL + 255) / 256, 256, 0, stream>>>(org, rec, pred1);

  // ---- upsampled outputs ----
  {
    long total = (long)BB * LL * 4 * OUT_R * OUT_R;
    upsample_a_kernel<<<(int)((total + 255) / 256), 256, 0, stream>>>(omc, out_a);
  }
  minsim_kernel<<<(BB * OUT_R * OUT_R + 255) / 256, 256, 0, stream>>>(mins, pred1, out_ms);
  diff_kernel<<<1, 32, 0, stream>>>(accum, out_diff);
}
